// ExtractingBaselineRnn_23278722744391
// MI455X (gfx1250) — compile-verified, matrix operand order checked by reading, not run on hardware
//
#include <hip/hip_runtime.h>
#include <hip/hip_bf16.h>

// ---------------------------------------------------------------------------
// LSTM (B=64, T=512, I=128, H=1024) + dense (H -> O=128) for gfx1250 (MI455X)
// bf16 WMMA (v_wmma_f32_16x16x32_bf16) for all GEMMs, fp32 cell state.
//   - Wk transposed to [4H x (I+H)] bf16 so B-fragments are contiguous b128s.
//   - One fused kernel per timestep: x-part K-loop (4 steps, unrolled) +
//     h-part K-loop (32 steps, two-stage software pipelined) + gates + c/h.
//   - Block covers 4 m-tiles x 2 n-tiles so waves needing the same Wk
//     B-fragment are co-resident on one WGP (WGP$ hits, not 4x L2 traffic).
//   - h written as bf16 into rnn_out[B,T,H]; next step reads it as A matrix.
//   - t==0 skips the h-part loop entirely (h_{-1} == 0).
// ---------------------------------------------------------------------------

#define B_ 64
#define T_ 512
#define I_ 128
#define H_ 1024
#define O_ 128
#define KCAT 1152   // I_ + H_
#define G4H 4096    // 4*H_

typedef __attribute__((ext_vector_type(16))) __bf16 v16bf;
typedef __attribute__((ext_vector_type(8)))  __bf16 v8bf;
typedef __attribute__((ext_vector_type(8)))  float  v8f;

__device__ __forceinline__ float sigmoidf_(float x) {
    return 1.0f / (1.0f + __expf(-x));
}

// Load a 16x32 (MxK) 16-bit WMMA fragment from a row-major matrix.
// Per CDNA5 ISA 7.12.2: lanes 0-15 hold row M=lane, K={k0..k0+7, k0+16..k0+23};
// lanes 16-31 hold row M=lane-16, K={k0+8..k0+15, k0+24..k0+31}.
// 'p' must already point at (row_for_this_lane, k0 + (lane>=16 ? 8 : 0)).
__device__ __forceinline__ v16bf load_frag16(const __bf16* __restrict__ p) {
    v8bf lo = *(const v8bf*)(p);
    v8bf hi = *(const v8bf*)(p + 16);
    v16bf r;
#pragma unroll
    for (int i = 0; i < 8; ++i) { r[i] = lo[i]; r[i + 8] = hi[i]; }
    return r;
}

#define WMMA_BF16(A, Bm, C) \
    __builtin_amdgcn_wmma_f32_16x16x32_bf16(false, (A), false, (Bm), (short)0, (C), false, false)

// --------------------------- prep kernels ----------------------------------

__global__ void kConvertX(const float* __restrict__ X, __bf16* __restrict__ Xbf, int n) {
    int i = blockIdx.x * blockDim.x + threadIdx.x;
    if (i < n) Xbf[i] = (__bf16)X[i];
}

// Wk [KCAT x 4H] row-major fp32 -> WkT [4H x KCAT] row-major bf16
__global__ void kTransWk(const float* __restrict__ Wk, __bf16* __restrict__ WkT) {
    int i = blockIdx.x * blockDim.x + threadIdx.x;
    if (i < KCAT * G4H) {
        int k = i / G4H;
        int n = i % G4H;
        WkT[(size_t)n * KCAT + k] = (__bf16)Wk[i];
    }
}

// Wd [H x O] fp32 -> WdT [O x H] bf16
__global__ void kTransWd(const float* __restrict__ Wd, __bf16* __restrict__ WdT) {
    int i = blockIdx.x * blockDim.x + threadIdx.x;
    if (i < H_ * O_) {
        int h = i / O_;
        int o = i % O_;
        WdT[(size_t)o * H_ + h] = (__bf16)Wd[i];
    }
}

__global__ void kZeroC(float* __restrict__ c, int n) {
    int i = blockIdx.x * blockDim.x + threadIdx.x;
    if (i < n) c[i] = 0.0f;
}

// --------------------------- fused LSTM step -------------------------------
// Grid: 32 blocks x 256 threads = 256 waves.
// Wave (block b, warp p) owns tile m0 = (p&3)*16, n0 = (b*2 + (p>>2))*16,
// for all 4 gates: the 4 waves with equal n0 (different m0) share identical
// Wk B-fragments and sit on the same WGP; the 2 waves with equal m0 share
// the A fragment.
__global__ void __launch_bounds__(256)
kLstmStep(const __bf16* __restrict__ Xbf,   // [B,T,I] bf16
          const __bf16* __restrict__ WkT,   // [4H x KCAT] bf16
          const float*  __restrict__ bias,  // [4H] fp32
          float*        __restrict__ c,     // [B x H] fp32 state
          __bf16*       __restrict__ rnn,   // [B,T,H] bf16 (h outputs)
          int t)
{
    const int lane  = threadIdx.x & 31;
    const int warp  = threadIdx.x >> 5;
    const int m0    = (warp & 3) << 4;                       // batch tile base
    const int n0    = ((blockIdx.x << 1) + (warp >> 2)) << 4; // hidden tile base
    const int lmod  = lane & 15;
    const int co    = (lane >> 4) << 3; // 0 or 8: lane K sub-offset

    v8f acc_i = {}, acc_j = {}, acc_f = {}, acc_o = {};

    const int rowm = m0 + lmod;  // this lane's A row (batch index)
    const __bf16* xrow = Xbf + ((size_t)rowm * T_ + t) * I_ + co;
    const __bf16* bi0  = WkT + (size_t)(0 * H_ + n0 + lmod) * KCAT + co;
    const __bf16* bj0  = WkT + (size_t)(1 * H_ + n0 + lmod) * KCAT + co;
    const __bf16* bf0  = WkT + (size_t)(2 * H_ + n0 + lmod) * KCAT + co;
    const __bf16* bo0  = WkT + (size_t)(3 * H_ + n0 + lmod) * KCAT + co;

    // ---- x_t part: K = 0..127, 4 steps, fully unrolled, branch-free ----
#pragma unroll
    for (int k0 = 0; k0 < I_; k0 += 32) {
        v16bf a  = load_frag16(xrow + k0);
        v16bf fi = load_frag16(bi0 + k0);
        v16bf fj = load_frag16(bj0 + k0);
        v16bf ff = load_frag16(bf0 + k0);
        v16bf fo = load_frag16(bo0 + k0);
        acc_i = WMMA_BF16(a, fi, acc_i);
        acc_j = WMMA_BF16(a, fj, acc_j);
        acc_f = WMMA_BF16(a, ff, acc_f);
        acc_o = WMMA_BF16(a, fo, acc_o);
    }

    // ---- h_{t-1} part: K = 128..1151, 32 steps, two-stage pipeline ----
    // Skipped entirely when t == 0 (h_{-1} == 0 contributes nothing).
    if (t > 0) {
        const __bf16* hrow = rnn + ((size_t)rowm * T_ + (t - 1)) * H_ + co;

        // Stage-A preload (k = 128).
        v16bf aA = load_frag16(hrow);
        v16bf iA = load_frag16(bi0 + 128);
        v16bf jA = load_frag16(bj0 + 128);
        v16bf fA = load_frag16(bf0 + 128);
        v16bf oA = load_frag16(bo0 + 128);

        // 16 iterations x 2 K-steps; final stage-A load over-reads 32 cols
        // (never consumed; workspace has a pad region after rnn).
        for (int k0 = 128; k0 < KCAT; k0 += 64) {
            // Load stage B (k0+32) while stage-A WMMAs execute.
            v16bf aB = load_frag16(hrow + (k0 + 32 - I_));
            v16bf iB = load_frag16(bi0 + k0 + 32);
            v16bf jB = load_frag16(bj0 + k0 + 32);
            v16bf fB = load_frag16(bf0 + k0 + 32);
            v16bf oB = load_frag16(bo0 + k0 + 32);

            acc_i = WMMA_BF16(aA, iA, acc_i);
            acc_j = WMMA_BF16(aA, jA, acc_j);
            acc_f = WMMA_BF16(aA, fA, acc_f);
            acc_o = WMMA_BF16(aA, oA, acc_o);

            // Load stage A (k0+64) while stage-B WMMAs execute.
            aA = load_frag16(hrow + (k0 + 64 - I_));
            iA = load_frag16(bi0 + k0 + 64);
            jA = load_frag16(bj0 + k0 + 64);
            fA = load_frag16(bf0 + k0 + 64);
            oA = load_frag16(bo0 + k0 + 64);

            acc_i = WMMA_BF16(aB, iB, acc_i);
            acc_j = WMMA_BF16(aB, jB, acc_j);
            acc_f = WMMA_BF16(aB, fB, acc_f);
            acc_o = WMMA_BF16(aB, oB, acc_o);
        }
    }

    // Epilogue: gate activations + cell/hidden update.
    // C/D layout: VGPR v, lane l -> M = m0 + v + 8*(l>=16), N = n0 + (l&15).
    const int n  = n0 + lmod;
    const float bbi = bias[0 * H_ + n];
    const float bbj = bias[1 * H_ + n];
    const float bbf = bias[2 * H_ + n] + 1.0f;  // forget bias
    const float bbo = bias[3 * H_ + n];
    const int mbase = m0 + ((lane >> 4) << 3);

#pragma unroll
    for (int v = 0; v < 8; ++v) {
        const int m = mbase + v;
        const size_t cidx = (size_t)m * H_ + n;
        float ig = sigmoidf_(acc_i[v] + bbi);
        float fg = sigmoidf_(acc_f[v] + bbf);
        float og = sigmoidf_(acc_o[v] + bbo);
        float g  = fmaxf(acc_j[v] + bbj, 0.0f);       // relu cell input
        float cn = fg * c[cidx] + ig * g;
        c[cidx]  = cn;
        float h  = og * fmaxf(cn, 0.0f);              // relu output act
        rnn[((size_t)m * T_ + t) * H_ + n] = (__bf16)h;
    }
}

// --------------------------- final dense -----------------------------------
// out[B*T x O] = rnn[B*T x H] @ Wd + bd ; 2048 M-tiles x 8 N-tiles.
// All 8 waves of a block share the same A rows (WGP$ hits); WdT (256 KB)
// is fully L2-resident.
__global__ void __launch_bounds__(256)
kDense(const __bf16* __restrict__ rnn,   // [B*T x H] bf16
       const __bf16* __restrict__ WdT,   // [O x H] bf16
       const float*  __restrict__ bd,    // [O]
       float*        __restrict__ out)   // [B*T x O] fp32
{
    const int lane = threadIdx.x & 31;
    const int w    = (blockIdx.x << 3) + (threadIdx.x >> 5);
    const int m0   = (w >> 3) << 4;
    const int n0   = (w & 7) << 4;
    const int lmod = lane & 15;
    const int co   = (lane >> 4) << 3;

    v8f acc = {};
    const __bf16* arow = rnn + (size_t)(m0 + lmod) * H_ + co;
    const __bf16* brow = WdT + (size_t)(n0 + lmod) * H_ + co;

    // Two-stage pipeline over K = 0..1023 (32 steps, 16 iterations).
    v16bf aA = load_frag16(arow);
    v16bf bA = load_frag16(brow);
    for (int k0 = 0; k0 < H_; k0 += 64) {
        v16bf aB = load_frag16(arow + k0 + 32);
        v16bf bB = load_frag16(brow + k0 + 32);
        acc = WMMA_BF16(aA, bA, acc);
        aA = load_frag16(arow + k0 + 64);   // over-reads on last iter (padded)
        bA = load_frag16(brow + k0 + 64);
        acc = WMMA_BF16(aB, bB, acc);
    }

    const int n = n0 + lmod;
    const float bb = bd[n];
    const int mbase = m0 + ((lane >> 4) << 3);
#pragma unroll
    for (int v = 0; v < 8; ++v) {
        out[(size_t)(mbase + v) * O_ + n] = acc[v] + bb;
    }
}

// --------------------------- launcher --------------------------------------

extern "C" void kernel_launch(void* const* d_in, const int* in_sizes, int n_in,
                              void* d_out, int out_size, void* d_ws, size_t ws_size,
                              hipStream_t stream) {
    const float* X  = (const float*)d_in[0];   // [B,T,I]
    const float* Wk = (const float*)d_in[1];   // [KCAT, 4H]
    const float* b  = (const float*)d_in[2];   // [4H]
    const float* Wd = (const float*)d_in[3];   // [H, O]
    const float* bd = (const float*)d_in[4];   // [O]
    float* out = (float*)d_out;                // [B,T,O]

    // Workspace layout (all 256B-aligned). Pipeline over-reads stay in-bounds
    // because a pad region follows the last live buffer.
    char* ws = (char*)d_ws;
    size_t off = 0;
    __bf16* Xbf = (__bf16*)(ws + off); off += (size_t)B_ * T_ * I_ * 2;       //  8.0 MiB
    __bf16* WkT = (__bf16*)(ws + off); off += (size_t)G4H * KCAT * 2;         //  9.0 MiB
    __bf16* WdT = (__bf16*)(ws + off); off += (size_t)O_ * H_ * 2;            //  0.25 MiB
    float*  c   = (float*) (ws + off); off += (size_t)B_ * H_ * 4;            //  0.25 MiB
    __bf16* rnn = (__bf16*)(ws + off); off += (size_t)B_ * T_ * H_ * 2;       // 64.0 MiB
    off += 4096;                                                              // over-read pad

    const int nX = B_ * T_ * I_;
    kConvertX<<<(nX + 255) / 256, 256, 0, stream>>>(X, Xbf, nX);
    kTransWk<<<(KCAT * G4H + 255) / 256, 256, 0, stream>>>(Wk, WkT);
    kTransWd<<<(H_ * O_ + 255) / 256, 256, 0, stream>>>(Wd, WdT);
    kZeroC<<<(B_ * H_ + 255) / 256, 256, 0, stream>>>(c, B_ * H_);

    // Sequential recurrence: one fused kernel per timestep.
    for (int t = 0; t < T_; ++t) {
        kLstmStep<<<32, 256, 0, stream>>>(Xbf, WkT, b, c, rnn, t);
    }

    // Final dense over all B*T rows.
    kDense<<<2048, 256, 0, stream>>>(rnn, WdT, bd, out);
}
